// SelfAttention_5463198400821
// MI455X (gfx1250) — matrix operand order repaired and checked by reading, not compiled, not run on hardware
//
#include <hip/hip_runtime.h>
#include <hip/hip_bf16.h>

// ---------------------------------------------------------------------------
// Self-attention (B=8, S=2048, D=1024, single head, no 1/sqrt(d) scale)
//   q,k,v = x @ W*.T + b*  ; scores = q k^T ; P = softmax(scores) ; out = P v
//   result = gamma * out + x      (gamma == 0 in the harness -> exact)
// bf16 WMMA (v_wmma_f32_16x16x32_bf16), fp32 accumulate.
// GEMMs: block tile 128x128 (4 waves), TDM (tensor_load_to_lds) double-
// buffered LDS staging (K stage 64 -> 32 WMMA per barrier round), wave tile
// 64x64 (16 acc, 512B LDS traffic per WMMA) via ds_load_b128 fragments.
// V transpose uses global_load_async_to_lds_b128 (ASYNCcnt path).
// ---------------------------------------------------------------------------

#define DEVI __device__ __forceinline__

typedef __bf16 bf16;
typedef __attribute__((ext_vector_type(16))) bf16  v16bf;
typedef __attribute__((ext_vector_type(8)))  bf16  v8bf;
typedef __attribute__((ext_vector_type(8)))  float v8f;
typedef unsigned int u32x4 __attribute__((ext_vector_type(4)));
typedef int i32x8 __attribute__((ext_vector_type(8)));
typedef int i32x4 __attribute__((ext_vector_type(4)));

constexpr int Bn = 8, Sn = 2048, Dn = 1024;
constexpr int LDP = 72;             // LDS row stride in elems (128B + 16B pad)
constexpr int STG_E = 128 * LDP;    // elements per LDS stage tile
constexpr int STG_B = STG_E * 2;    // bytes per LDS stage tile (18,432)

DEVI int lane_id() { return (int)(threadIdx.x & 31u); }

DEVI unsigned short f32_to_bf16u(float f) {
  union { float f; unsigned int u; } a; a.f = f;
  unsigned int u = a.u;
  u = (u + 0x7FFFu + ((u >> 16) & 1u)) >> 16;   // round-to-nearest-even
  return (unsigned short)u;
}
DEVI float bf16u_to_f32(unsigned int u) {
  union { unsigned int u; float f; } a; a.u = u << 16;
  return a.f;
}
DEVI unsigned lds_off32(const void* p) {   // workgroup-relative LDS byte offset
  return (unsigned)(unsigned long long)p;
}

// A fragment: 16x32 (MxK) bf16, row-major source, K contiguous, stride ld.
DEVI v16bf load_a16x32(const unsigned short* p, int ld) {
  const int l = lane_id();
  const bf16* q = (const bf16*)(p + (size_t)(l & 15) * ld + ((l >> 4) << 3));
  v8bf lo = *(const v8bf*)(q);        // K = ksel + 0..7
  v8bf hi = *(const v8bf*)(q + 16);   // K = ksel + 16..23
  return __builtin_shufflevector(lo, hi, 0,1,2,3,4,5,6,7,8,9,10,11,12,13,14,15);
}

// B fragment: 32x16 (KxN) bf16 from [N rows x K] row-major source (NT GEMM).
DEVI v16bf load_b32x16(const unsigned short* p, int ld) {
  const int l = lane_id();
  const bf16* q = (const bf16*)(p + (size_t)(l & 15) * ld + ((l >> 4) << 3));
  v8bf lo = *(const v8bf*)(q);
  v8bf hi = *(const v8bf*)(q + 16);
  return __builtin_shufflevector(lo, hi, 0,1,2,3,4,5,6,7,8,9,10,11,12,13,14,15);
}

DEVI v8f wmma_bf16(v16bf a, v16bf b, v8f c) {
  return __builtin_amdgcn_wmma_f32_16x16x32_bf16(
      false, a, false, b, (short)0, c, false, false);
}

// C/D 16x16 f32: lane = col (lane&15); lanes>=16 hold rows 8..15.
DEVI void store_bf16_tile(unsigned short* p, int ld, const v8f c) {
  const int l = lane_id();
  const int col = l & 15;
  const int r0 = (l >> 4) * 8;
#pragma unroll
  for (int i = 0; i < 8; ++i)
    p[(size_t)(r0 + i) * ld + col] = f32_to_bf16u(c[i]);
}

DEVI void store_f32_ep(float* o, const float* xr, int ld, const v8f c, float g) {
  const int l = lane_id();
  const int col = l & 15;
  const int r0 = (l >> 4) * 8;
#pragma unroll
  for (int i = 0; i < 8; ++i) {
    const size_t idx = (size_t)(r0 + i) * ld + col;
    o[idx] = g * c[i] + xr[idx];
  }
}

// ------------------------- TDM descriptor helpers ---------------------------
// D# group1 for a 2D bf16 tile: 128 rows x 64 elems (128B), LDS padding 16B
// after every 128B (pad_interval=4 -> 32 dwords, pad_amount=3 -> 4 dwords),
// giving LDS row stride 144B = LDP elements. data_size=2B.
DEVI i32x8 tdm_desc_g1(unsigned ld) {
  i32x8 g1;
  g1[0] = (1 << 16) | (1 << 20) | (4 << 22) | (3 << 25);
  g1[1] = (int)(64u << 16);     // tensor_dim0 = 64 (bits 79:48, low part)
  g1[2] = (int)(128u << 16);    // tensor_dim0 hi | tensor_dim1 = 128 (low16)
  g1[3] = (int)(64u << 16);     // tensor_dim1 hi | tile_dim0 = 64
  g1[4] = 128;                  // tile_dim1 = 128 (tile_dim2 = 0)
  g1[5] = (int)ld;              // tensor_dim0_stride (row stride, elements)
  g1[6] = 0;
  g1[7] = 0;
  return g1;
}

DEVI void tdm_issue(unsigned ldsoff, unsigned long long ga, i32x8 g1) {
  u32x4 g0;
  g0.x = 1u;                                    // count = 1 (user D#)
  g0.y = ldsoff;                                // LDS byte address
  g0.z = (unsigned)ga;                          // global_addr[31:0]
  g0.w = (unsigned)((ga >> 32) & 0x01FFFFFFu) | (2u << 30);   // addr hi | type=2
  const i32x4 z4 = {0, 0, 0, 0};
  const i32x8 z8 = {0, 0, 0, 0, 0, 0, 0, 0};
  __builtin_amdgcn_tensor_load_to_lds(g0, g1, z4, z4, z8, 0);
}

// TDM double-buffered block GEMM: C[128x128] += A[128xK] * B[128xK]^T.
// Ablk/Bblk point at the block's first row; ld = row stride = K.
// 4 waves; wave computes 64x64: rows (wave>>1)*64, cols (wave&1)*64.
DEVI void gemm_block_lds(const unsigned short* Ablk, const unsigned short* Bblk,
                         int ld, int K, int wave,
                         unsigned short* shA, unsigned short* shB,
                         v8f (&acc)[4][4]) {
  const int NS = K >> 6;                       // 64-wide K stages
  const unsigned baseA = lds_off32(shA);
  const unsigned baseB = lds_off32(shB);
  const i32x8 g1 = tdm_desc_g1((unsigned)ld);
  const unsigned long long gaA = (unsigned long long)Ablk;
  const unsigned long long gaB = (unsigned long long)Bblk;

  if (wave == 0) {                             // prime stage 0
    tdm_issue(baseA, gaA, g1);
    tdm_issue(baseB, gaB, g1);
  }
  for (int s = 0; s < NS; ++s) {
    const int p = s & 1;
    if (wave == 0) __builtin_amdgcn_s_wait_tensorcnt(0);
    __syncthreads();                           // stage s resident for all waves
    if (wave == 0 && s + 1 < NS) {             // overlap DMA of s+1 w/ compute
      tdm_issue(baseA + (unsigned)(p ^ 1) * STG_B, gaA + (size_t)(s + 1) * 128, g1);
      tdm_issue(baseB + (unsigned)(p ^ 1) * STG_B, gaB + (size_t)(s + 1) * 128, g1);
    }
    const unsigned short* As = shA + p * STG_E + ((wave >> 1) * 64) * LDP;
    const unsigned short* Bs = shB + p * STG_E + ((wave & 1) * 64) * LDP;
#pragma unroll
    for (int kk = 0; kk < 64; kk += 32) {
      v16bf a[4], b[4];
#pragma unroll
      for (int r = 0; r < 4; ++r) {
        a[r] = load_a16x32(As + r * 16 * LDP + kk, LDP);
        b[r] = load_b32x16(Bs + r * 16 * LDP + kk, LDP);
      }
#pragma unroll
      for (int i = 0; i < 4; ++i)
#pragma unroll
        for (int j = 0; j < 4; ++j)
          acc[i][j] = wmma_bf16(a[i], b[j], acc[i][j]);
    }
  }
}

#define GEMM_SHARED_DECL                                                      \
  __shared__ unsigned short shA[2 * STG_E];                                   \
  __shared__ unsigned short shB[2 * STG_E];

// --------------------------- kernels ---------------------------------------

__global__ void __launch_bounds__(256)
convert_f32_to_bf16_kernel(const float* __restrict__ src,
                           unsigned short* __restrict__ dst, int n4) {
  const int i = blockIdx.x * blockDim.x + threadIdx.x;
  if (i >= n4) return;
  const float4 f = ((const float4*)src)[i];
  ushort4 o;
  o.x = f32_to_bf16u(f.x); o.y = f32_to_bf16u(f.y);
  o.z = f32_to_bf16u(f.z); o.w = f32_to_bf16u(f.w);
  ((ushort4*)dst)[i] = o;
}

// Q/K/V projection: [16384 x 1024] = xb @ W^T + b (NT). Block tile 128x128.
__global__ void __launch_bounds__(128)
proj_gemm_kernel(const unsigned short* __restrict__ xb,
                 const unsigned short* __restrict__ wb,
                 const float* __restrict__ bq, const float* __restrict__ bk,
                 const float* __restrict__ bv,
                 unsigned short* __restrict__ Qb,
                 unsigned short* __restrict__ Kb,
                 unsigned short* __restrict__ Vb) {
  GEMM_SHARED_DECL
  const int z = blockIdx.z;
  const unsigned short* W = wb + (size_t)z * Dn * Dn;
  const float* bias = (z == 0) ? bq : (z == 1) ? bk : bv;
  unsigned short* out = (z == 0) ? Qb : (z == 1) ? Kb : Vb;

  const int wave = threadIdx.x >> 5;
  const int m0 = blockIdx.x * 128;
  const int n0 = blockIdx.y * 128;

  v8f acc[4][4] = {};
  gemm_block_lds(xb + (size_t)m0 * Dn, W + (size_t)n0 * Dn, Dn, Dn, wave,
                 shA, shB, acc);

  const int mw = m0 + (wave >> 1) * 64;
  const int nw = n0 + (wave & 1) * 64;
  const int col = lane_id() & 15;
#pragma unroll
  for (int j = 0; j < 4; ++j) {
    const float bj = bias[nw + 16 * j + col];
#pragma unroll
    for (int i = 0; i < 4; ++i)
#pragma unroll
      for (int e = 0; e < 8; ++e) acc[i][j][e] += bj;
  }
#pragma unroll
  for (int mi = 0; mi < 4; ++mi)
#pragma unroll
    for (int j = 0; j < 4; ++j)
      store_bf16_tile(out + (size_t)(mw + 16 * mi) * Dn + nw + 16 * j, Dn,
                      acc[mi][j]);
}

// V [B,S,D] -> Vt [B,D,S]. 32(s) x 64(d) tile staged in LDS via
// global_load_async_to_lds_b128 (ASYNCcnt), then coalesced b128 stores.
__global__ void __launch_bounds__(256)
transpose_kernel(const unsigned short* __restrict__ V,
                 unsigned short* __restrict__ Vt) {
  __shared__ unsigned short tile[32][72];   // row stride 144B (pad, 16B align)
  const int b = blockIdx.z;
  const int s0 = blockIdx.x * 32;
  const int d0 = blockIdx.y * 64;
  const unsigned short* src = V + (size_t)b * Sn * Dn;
  unsigned short* dst = Vt + (size_t)b * Dn * Sn;

  const int t = threadIdx.x;
  {
    const int row = t >> 3;          // 0..31  (s index within tile)
    const int ch  = t & 7;           // 0..7   (16B chunk along d)
    const unsigned ldsoff = lds_off32(&tile[0][0]) + row * 144u + ch * 16u;
    const unsigned voff =
        (unsigned)(((size_t)(s0 + row) * Dn + d0 + ch * 8) * 2);
    asm volatile("global_load_async_to_lds_b128 %0, %1, %2"
                 :: "v"(ldsoff), "v"(voff), "s"(src) : "memory");
  }
  asm volatile("s_wait_asynccnt 0" ::: "memory");
  __syncthreads();
  {
    const int drow = t >> 2;         // 0..63
    const int ch   = t & 3;          // 0..3
    union { unsigned short u[8]; uint4 v; } pk;
#pragma unroll
    for (int i = 0; i < 8; ++i) pk.u[i] = tile[ch * 8 + i][drow];
    *(uint4*)(dst + (size_t)(d0 + drow) * Sn + s0 + ch * 8) = pk.v;
  }
}

// scores[b] = Q[b] @ K[b]^T : M=N=2048, K=1024 (NT), bf16 out.
__global__ void __launch_bounds__(128)
scores_gemm_kernel(const unsigned short* __restrict__ Qb,
                   const unsigned short* __restrict__ Kb,
                   unsigned short* __restrict__ Sc) {
  GEMM_SHARED_DECL
  const int b = blockIdx.z;
  const unsigned short* A = Qb + (size_t)b * Sn * Dn;
  const unsigned short* Bm = Kb + (size_t)b * Sn * Dn;
  unsigned short* out = Sc + (size_t)b * Sn * Sn;

  const int wave = threadIdx.x >> 5;
  const int m0 = blockIdx.x * 128;
  const int n0 = blockIdx.y * 128;

  v8f acc[4][4] = {};
  gemm_block_lds(A + (size_t)m0 * Dn, Bm + (size_t)n0 * Dn, Dn, Dn, wave,
                 shA, shB, acc);

  const int mw = m0 + (wave >> 1) * 64;
  const int nw = n0 + (wave & 1) * 64;
#pragma unroll
  for (int mi = 0; mi < 4; ++mi)
#pragma unroll
    for (int j = 0; j < 4; ++j)
      store_bf16_tile(out + (size_t)(mw + 16 * mi) * Sn + nw + 16 * j, Sn,
                      acc[mi][j]);
}

// In-place row softmax over 2048 bf16; one 256-thread block per row.
__global__ void __launch_bounds__(256)
softmax_kernel(unsigned short* __restrict__ scores) {
  __shared__ float red[8];
  unsigned short* p = scores + (size_t)blockIdx.x * Sn;
  const int t = threadIdx.x;

  const uint4 raw = *(const uint4*)(p + t * 8);
  unsigned int wr[4] = {raw.x, raw.y, raw.z, raw.w};
  float v[8];
#pragma unroll
  for (int i = 0; i < 4; ++i) {
    v[2 * i]     = bf16u_to_f32(wr[i] & 0xFFFFu);
    v[2 * i + 1] = bf16u_to_f32(wr[i] >> 16);
  }
  float mx = v[0];
#pragma unroll
  for (int i = 1; i < 8; ++i) mx = fmaxf(mx, v[i]);
#pragma unroll
  for (int o = 16; o > 0; o >>= 1) mx = fmaxf(mx, __shfl_xor(mx, o, 32));
  if ((t & 31) == 0) red[t >> 5] = mx;
  __syncthreads();
  float gmx = red[0];
#pragma unroll
  for (int j = 1; j < 8; ++j) gmx = fmaxf(gmx, red[j]);
  __syncthreads();

  float s = 0.f;
#pragma unroll
  for (int i = 0; i < 8; ++i) { v[i] = __expf(v[i] - gmx); s += v[i]; }
#pragma unroll
  for (int o = 16; o > 0; o >>= 1) s += __shfl_xor(s, o, 32);
  if ((t & 31) == 0) red[t >> 5] = s;
  __syncthreads();
  float gs = red[0];
#pragma unroll
  for (int j = 1; j < 8; ++j) gs += red[j];
  const float inv = 1.0f / gs;

  uint4 ow;
  ow.x = (unsigned int)f32_to_bf16u(v[0] * inv) | ((unsigned int)f32_to_bf16u(v[1] * inv) << 16);
  ow.y = (unsigned int)f32_to_bf16u(v[2] * inv) | ((unsigned int)f32_to_bf16u(v[3] * inv) << 16);
  ow.z = (unsigned int)f32_to_bf16u(v[4] * inv) | ((unsigned int)f32_to_bf16u(v[5] * inv) << 16);
  ow.w = (unsigned int)f32_to_bf16u(v[6] * inv) | ((unsigned int)f32_to_bf16u(v[7] * inv) << 16);
  *(uint4*)(p + t * 8) = ow;
}

// out[b] = P[b] @ Vt[b]^T : M=2048, N=1024, K=2048 (NT), fused gamma*c + x.
__global__ void __launch_bounds__(128)
out_gemm_kernel(const unsigned short* __restrict__ Pm,
                const unsigned short* __restrict__ Vt,
                const float* __restrict__ x, const float* __restrict__ gamma_p,
                float* __restrict__ out) {
  GEMM_SHARED_DECL
  const int b = blockIdx.z;
  const unsigned short* A = Pm + (size_t)b * Sn * Sn;
  const unsigned short* Bm = Vt + (size_t)b * Dn * Sn;

  const int wave = threadIdx.x >> 5;
  const int m0 = blockIdx.x * 128;
  const int n0 = blockIdx.y * 128;

  v8f acc[4][4] = {};
  gemm_block_lds(A + (size_t)m0 * Sn, Bm + (size_t)n0 * Sn, Sn, Sn, wave,
                 shA, shB, acc);

  const int mw = m0 + (wave >> 1) * 64;
  const int nw = n0 + (wave & 1) * 64;
  const float g = gamma_p[0];
  const float* xs = x + ((size_t)b * Sn + mw) * Dn;
  float* os = out + ((size_t)b * Sn + mw) * Dn;
#pragma unroll
  for (int mi = 0; mi < 4; ++mi)
#pragma unroll
    for (int j = 0; j < 4; ++j) {
      const size_t o = (size_t)(16 * mi) * Dn + nw + 16 * j;
      store_f32_ep(os + o, xs + o, Dn, acc[mi][j], g);
    }
}

// --------------------------- launcher ---------------------------------------

extern "C" void kernel_launch(void* const* d_in, const int* in_sizes, int n_in,
                              void* d_out, int out_size, void* d_ws, size_t ws_size,
                              hipStream_t stream) {
  (void)in_sizes; (void)n_in; (void)out_size; (void)ws_size;

  const float* x     = (const float*)d_in[0];
  const float* Wq    = (const float*)d_in[1];
  const float* bq    = (const float*)d_in[2];
  const float* Wk    = (const float*)d_in[3];
  const float* bk    = (const float*)d_in[4];
  const float* Wv    = (const float*)d_in[5];
  const float* bv    = (const float*)d_in[6];
  const float* gamma = (const float*)d_in[7];
  float* out = (float*)d_out;

  const size_t XE = (size_t)Bn * Sn * Dn;   // 16,777,216
  const size_t WE = (size_t)Dn * Dn;        //  1,048,576
  const size_t SE = (size_t)Bn * Sn * Sn;   // 33,554,432

  char* ws = (char*)d_ws;
  size_t off = 0;
  unsigned short* xb = (unsigned short*)(ws + off); off += XE * 2;
  unsigned short* wb = (unsigned short*)(ws + off); off += 3 * WE * 2;
  unsigned short* Qb = (unsigned short*)(ws + off); off += XE * 2;
  unsigned short* Kb = (unsigned short*)(ws + off); off += XE * 2;
  unsigned short* Vb = (unsigned short*)(ws + off); off += XE * 2;
  unsigned short* Vt = (unsigned short*)(ws + off); off += XE * 2;
  unsigned short* Sc = (unsigned short*)(ws + off); off += SE * 2;  // ~241 MB

  {
    const int n4 = (int)(XE / 4);
    convert_f32_to_bf16_kernel<<<(n4 + 255) / 256, 256, 0, stream>>>(x, xb, n4);
    const int w4 = (int)(WE / 4);
    convert_f32_to_bf16_kernel<<<(w4 + 255) / 256, 256, 0, stream>>>(Wq, wb, w4);
    convert_f32_to_bf16_kernel<<<(w4 + 255) / 256, 256, 0, stream>>>(Wk, wb + WE, w4);
    convert_f32_to_bf16_kernel<<<(w4 + 255) / 256, 256, 0, stream>>>(Wv, wb + 2 * WE, w4);
  }

  proj_gemm_kernel<<<dim3((Bn * Sn) / 128, Dn / 128, 3), 128, 0, stream>>>(
      xb, wb, bq, bk, bv, Qb, Kb, Vb);

  transpose_kernel<<<dim3(Sn / 32, Dn / 64, Bn), 256, 0, stream>>>(Vb, Vt);

  scores_gemm_kernel<<<dim3(Sn / 128, Sn / 128, Bn), 128, 0, stream>>>(Qb, Kb, Sc);

  softmax_kernel<<<Bn * Sn, 256, 0, stream>>>(Sc);

  out_gemm_kernel<<<dim3(Sn / 128, Dn / 128, Bn), 128, 0, stream>>>(
      Sc, Vt, x, gamma, out);
}
